// GAT_90460601188538
// MI455X (gfx1250) — matrix operand order repaired and checked by reading, not compiled, not run on hardware
//
#include <hip/hip_runtime.h>
#include <hip/hip_bf16.h>
#include <math.h>
#include <stdint.h>

typedef __attribute__((ext_vector_type(16))) _Float16 v16h;
typedef __attribute__((ext_vector_type(8)))  _Float16 v8h;
typedef __attribute__((ext_vector_type(8)))  float    v8f;

// exact pointee type expected by __builtin_amdgcn_global_load_async_to_lds_b128
typedef int v4i_gcc __attribute__((vector_size(16)));
typedef __attribute__((address_space(1))) v4i_gcc* gv4p;   // global (__device__)
typedef __attribute__((address_space(3))) v4i_gcc* lv4p;   // LDS (__shared__)

#define NEG_SLOPE 0.2f

#if defined(__has_builtin)
#if __has_builtin(__builtin_amdgcn_global_load_async_to_lds_b128)
#define HAVE_ASYNC_LDS 1
#endif
#endif

static __device__ __forceinline__ void atomicMaxFloat(float* addr, float val) {
    val = val + 0.0f;  // canonicalize -0 -> +0
    if (val >= 0.0f) {
        atomicMax((int*)addr, __float_as_int(val));
    } else {
        atomicMin((unsigned int*)addr, __float_as_uint(val));
    }
}

// ---------------------------------------------------------------------------
// Utility kernels
// ---------------------------------------------------------------------------
__global__ void fill_f32_kernel(float* __restrict__ p, float val, long long n) {
    long long i = (long long)blockIdx.x * blockDim.x + threadIdx.x;
    if (i < n) p[i] = val;
}

__global__ void cvt_f32_f16_kernel(const float* __restrict__ src,
                                   _Float16* __restrict__ dst, long long n) {
    long long i = (long long)blockIdx.x * blockDim.x + threadIdx.x;
    if (i < n) dst[i] = (_Float16)src[i];
}

// W [K, Cin] f32  ->  WT [Cpad, K] f16 (transposed, zero-padded columns)
__global__ void transpose_cvt_kernel(const float* __restrict__ W,
                                     _Float16* __restrict__ WT,
                                     int K, int Cin, int Cpad) {
    int i = blockIdx.x * blockDim.x + threadIdx.x;
    if (i >= Cpad * K) return;
    int c = i / K, k = i % K;
    WT[i] = (c < Cin) ? (_Float16)W[(size_t)k * Cin + c] : (_Float16)0.0f;
}

// ---------------------------------------------------------------------------
// WMMA GEMM: C[M,NCOL] = A[M,K] * B[K,NCOL], f16 in, f32 acc.
// BT is the transposed weight: BT[NCOL][K].
// Block = 256 threads (8 waves). The block's 16-row A tile is staged in LDS
// via async global->LDS (ASYNCcnt) when available; each wave then computes
// one or more 16x16 column tiles reading fragments from LDS (ds_load_b128)
// and BT (global_load_b128).  Fragment layouts per CDNA5 ISA 7.12.2.
// ---------------------------------------------------------------------------
template <int K, int NCOL>
__global__ void gemm_wmma_f16_kernel(const _Float16* __restrict__ A,
                                     const _Float16* __restrict__ BT,
                                     float* __restrict__ C, int M) {
    __shared__ __align__(16) _Float16 As[16 * K];

    const int lane  = threadIdx.x & 31;
    const int wave  = threadIdx.x >> 5;
    const int mtile = blockIdx.x;
    const int l15   = lane & 15;
    const int khalf = lane >> 4;

    // ---- stage A tile [16 x K] into LDS (16B chunks, one per thread) ----
    const int CHUNKS = 2 * K;          // (16*K*2)/16
    const int CPR    = K / 8;          // chunks per row
    for (int ch = threadIdx.x; ch < CHUNKS; ch += 256) {
        int row  = ch / CPR;
        int seg  = ch % CPR;
        int grow = mtile * 16 + row;
        if (grow >= M) grow = M - 1;
        const _Float16* gsrc = A + (size_t)grow * K + seg * 8;
        _Float16*       ldst = &As[row * K + seg * 8];
#ifdef HAVE_ASYNC_LDS
        __builtin_amdgcn_global_load_async_to_lds_b128(
            (gv4p)(uintptr_t)gsrc, (lv4p)(unsigned)(uintptr_t)ldst, 0, 0);
#else
        *(v8h*)ldst = *(const v8h*)gsrc;
#endif
    }
#ifdef HAVE_ASYNC_LDS
#if __has_builtin(__builtin_amdgcn_s_wait_asynccnt)
    __builtin_amdgcn_s_wait_asynccnt(0);
#else
    asm volatile("s_wait_asynccnt 0" ::: "memory");
#endif
#endif
    __syncthreads();

    const _Float16* arow = &As[l15 * K];
    const bool fullM = (mtile * 16 + 16 <= M);

    // each wave walks column tiles wave, wave+8, ... (wave-uniform loop)
    for (int nt = wave; nt * 16 < NCOL; nt += 8) {
        const int col = nt * 16 + l15;
        const _Float16* btcol = BT + (size_t)col * K;
        v8f acc = {};
#pragma unroll
        for (int ks = 0; ks < K / 32; ++ks) {
            const int kb = ks * 32;
            // A 16x32: lane row = l15; halves K = 8*khalf..+7 then 16+8*khalf..+7
            v8h alo = *(const v8h*)(arow + kb + 8 * khalf);
            v8h ahi = *(const v8h*)(arow + kb + 16 + 8 * khalf);
            // B 32x16: lane col = l15; halves K = 8*khalf..+7 then 16+8*khalf..+7 (from BT row)
            v8h blo = *(const v8h*)(btcol + kb + 8 * khalf);
            v8h bhi = *(const v8h*)(btcol + kb + 16 + 8 * khalf);
            v16h a = __builtin_shufflevector(alo, ahi, 0, 1, 2, 3, 4, 5, 6, 7,
                                             8, 9, 10, 11, 12, 13, 14, 15);
            v16h b = __builtin_shufflevector(blo, bhi, 0, 1, 2, 3, 4, 5, 6, 7,
                                             8, 9, 10, 11, 12, 13, 14, 15);
            acc = __builtin_amdgcn_wmma_f32_16x16x32_f16(
                      false, a, false, b, (short)0, acc, false, false);
        }
        // C/D: VGPR r -> (row = r + 8*khalf, col = l15)
        const int rbase = mtile * 16 + 8 * khalf;
        float* cp = C + (size_t)rbase * NCOL + col;
        if (fullM) {
#pragma unroll
            for (int r = 0; r < 8; ++r) cp[r * NCOL] = acc[r];
        } else {
#pragma unroll
            for (int r = 0; r < 8; ++r)
                if (rbase + r < M) cp[r * NCOL] = acc[r];
        }
    }
}

// ---------------------------------------------------------------------------
// Attention projections: a_s[n,h] = sum_c xw[n, h*C+c]*att_src[h,c]  (same a_d)
// ---------------------------------------------------------------------------
__global__ void att_dot_kernel(const float* __restrict__ xw,
                               const float* __restrict__ att_s,
                               const float* __restrict__ att_d,
                               float* __restrict__ a_s, float* __restrict__ a_d,
                               int N, int H, int C, int ld) {
    int idx = blockIdx.x * blockDim.x + threadIdx.x;
    if (idx >= N * H) return;
    int n = idx / H, h = idx % H;
    const float* row = xw + (size_t)n * ld + h * C;
    const float* s = att_s + h * C;
    const float* d = att_d + h * C;
    float accs = 0.f, accd = 0.f;
    for (int c = 0; c < C; ++c) {
        float v = row[c];
        accs += v * s[c];
        accd += v * d[c];
    }
    a_s[idx] = accs;
    a_d[idx] = accd;
}

// ---------------------------------------------------------------------------
// Edge passes.  Edges e < E from edge_index (int64 [2,E]); e >= E self loops.
// ---------------------------------------------------------------------------
__global__ void edge_max_kernel(const long long* __restrict__ ei,
                                const float* __restrict__ a_s,
                                const float* __restrict__ a_d,
                                float* __restrict__ m,
                                int E, int ET, int H) {
    int e = blockIdx.x * blockDim.x + threadIdx.x;
    if (e >= ET) return;
    int s, d;
    if (e < E) { s = (int)ei[e]; d = (int)ei[E + e]; } else { s = d = e - E; }
    for (int h = 0; h < H; ++h) {
        float v = a_s[s * H + h] + a_d[d * H + h];
        v = (v > 0.f) ? v : NEG_SLOPE * v;
        atomicMaxFloat(&m[d * H + h], v);
    }
}

__global__ void edge_exp_kernel(const long long* __restrict__ ei,
                                const float* __restrict__ a_s,
                                const float* __restrict__ a_d,
                                const float* __restrict__ m,
                                float* __restrict__ expe,
                                float* __restrict__ den,
                                int E, int ET, int H) {
    int e = blockIdx.x * blockDim.x + threadIdx.x;
    if (e >= ET) return;
    int s, d;
    if (e < E) { s = (int)ei[e]; d = (int)ei[E + e]; } else { s = d = e - E; }
    for (int h = 0; h < H; ++h) {
        float v = a_s[s * H + h] + a_d[d * H + h];
        v = (v > 0.f) ? v : NEG_SLOPE * v;
        float ex = __expf(v - m[d * H + h]);
        expe[(size_t)e * H + h] = ex;
        atomicAdd(&den[d * H + h], ex);
    }
}

// one edge per block; thread handles 4 channels (float4 b128 gather)
__global__ void edge_agg_kernel(const long long* __restrict__ ei,
                                const float* __restrict__ expe,
                                const float* __restrict__ den,
                                const float* __restrict__ feat,  // [N, ldS]
                                float* __restrict__ agg,         // [N, ldD]
                                int E, int H, int C, int ldS, int ldD) {
    int e  = blockIdx.x;
    int c4 = threadIdx.x;
    if (c4 * 4 >= H * C) return;
    int s, d;
    if (e < E) { s = (int)ei[e]; d = (int)ei[E + e]; } else { s = d = e - E; }
    int h = (c4 * 4) / C;
    float alpha = expe[(size_t)e * H + h] / den[d * H + h];
    const float4 f = *(const float4*)&feat[(size_t)s * ldS + 4 * c4];
    float* ap = &agg[(size_t)d * ldD + 4 * c4];
    atomicAdd(ap + 0, alpha * f.x);
    atomicAdd(ap + 1, alpha * f.y);
    atomicAdd(ap + 2, alpha * f.z);
    atomicAdd(ap + 3, alpha * f.w);
}

// h = elu(agg + b), emitted directly as f16 for the next WMMA GEMM
__global__ void elu_bias_f16_kernel(const float* __restrict__ agg,
                                    const float* __restrict__ b,
                                    _Float16* __restrict__ hh,
                                    long long total, int ld) {
    long long i = (long long)blockIdx.x * blockDim.x + threadIdx.x;
    if (i >= total) return;
    float v = agg[i] + b[i % ld];
    v = (v > 0.f) ? v : (__expf(v) - 1.0f);
    hh[i] = (_Float16)v;
}

// out[n,:] = log_softmax(agg2[n,:] + b2)
__global__ void log_softmax_kernel(const float* __restrict__ agg,
                                   const float* __restrict__ b,
                                   float* __restrict__ out, int N, int C) {
    int n = blockIdx.x * blockDim.x + threadIdx.x;
    if (n >= N) return;
    float v[64];
    float mx = -INFINITY;
    for (int c = 0; c < C; ++c) {
        float t = agg[(size_t)n * C + c] + b[c];
        v[c] = t;
        mx = fmaxf(mx, t);
    }
    float sum = 0.f;
    for (int c = 0; c < C; ++c) sum += __expf(v[c] - mx);
    float lse = mx + __logf(sum);
    for (int c = 0; c < C; ++c) out[(size_t)n * C + c] = v[c] - lse;
}

// ---------------------------------------------------------------------------
// Host launcher
// ---------------------------------------------------------------------------
static inline long long cdivll(long long a, long long b) { return (a + b - 1) / b; }

extern "C" void kernel_launch(void* const* d_in, const int* in_sizes, int n_in,
                              void* d_out, int out_size, void* d_ws, size_t ws_size,
                              hipStream_t stream) {
    const float*     x    = (const float*)d_in[0];
    const long long* ei   = (const long long*)d_in[1];
    const float*     W1   = (const float*)d_in[2];
    const float*     as1v = (const float*)d_in[3];
    const float*     ad1v = (const float*)d_in[4];
    const float*     b1   = (const float*)d_in[5];
    const float*     W2   = (const float*)d_in[6];
    const float*     as2v = (const float*)d_in[7];
    const float*     ad2v = (const float*)d_in[8];
    const float*     b2   = (const float*)d_in[9];
    float* out = (float*)d_out;

    const int F_IN = 128;
    const int H1 = 4, C1 = 64;
    const int F_MID = H1 * C1;           // 256
    const int C2 = in_sizes[9];          // 40
    const int C2P = 48;                  // padded to 3 WMMA column tiles
    const int N  = in_sizes[0] / F_IN;
    const int E  = in_sizes[1] / 2;
    const int ET = E + N;                // with self loops

    // ---- workspace bump allocator (256B aligned) ----
    char* base = (char*)d_ws;
    size_t off = 0;
    auto alloc = [&](size_t bytes) -> void* {
        void* p = base + off;
        off += (bytes + 255) & ~(size_t)255;
        return p;
    };
    _Float16* xh    = (_Float16*)alloc((size_t)N * F_IN * 2);
    _Float16* w1t   = (_Float16*)alloc((size_t)F_MID * F_IN * 2);   // [256][128]
    float*    xw1   = (float*)   alloc((size_t)N * F_MID * 4);
    float*    a_s1  = (float*)   alloc((size_t)N * H1 * 4);
    float*    a_d1  = (float*)   alloc((size_t)N * H1 * 4);
    float*    m1    = (float*)   alloc((size_t)N * H1 * 4);
    float*    den1  = (float*)   alloc((size_t)N * H1 * 4);
    float*    expe1 = (float*)   alloc((size_t)ET * H1 * 4);
    float*    agg1  = (float*)   alloc((size_t)N * F_MID * 4);
    _Float16* hh    = (_Float16*)alloc((size_t)N * F_MID * 2);
    _Float16* w2t   = (_Float16*)alloc((size_t)C2P * F_MID * 2);    // [48][256]
    float*    hw2   = (float*)   alloc((size_t)N * C2P * 4);
    float*    a_s2  = (float*)   alloc((size_t)N * 4);
    float*    a_d2  = (float*)   alloc((size_t)N * 4);
    float*    m2    = (float*)   alloc((size_t)N * 4);
    float*    den2  = (float*)   alloc((size_t)N * 4);
    float*    expe2 = (float*)   alloc((size_t)ET * 4);
    float*    agg2  = (float*)   alloc((size_t)N * C2 * 4);

    const int T = 256;

    // ===================== Layer 1 =====================
    cvt_f32_f16_kernel<<<cdivll((long long)N * F_IN, T), T, 0, stream>>>(
        x, xh, (long long)N * F_IN);
    transpose_cvt_kernel<<<cdivll((long long)F_MID * F_IN, T), T, 0, stream>>>(
        W1, w1t, F_IN, F_MID, F_MID);

    gemm_wmma_f16_kernel<128, 256><<<(N + 15) / 16, 256, 0, stream>>>(
        xh, w1t, xw1, N);

    att_dot_kernel<<<cdivll((long long)N * H1, T), T, 0, stream>>>(
        xw1, as1v, ad1v, a_s1, a_d1, N, H1, C1, F_MID);

    fill_f32_kernel<<<cdivll((long long)N * H1, T), T, 0, stream>>>(m1, -INFINITY, (long long)N * H1);
    fill_f32_kernel<<<cdivll((long long)N * H1, T), T, 0, stream>>>(den1, 0.f, (long long)N * H1);
    fill_f32_kernel<<<cdivll((long long)N * F_MID, T), T, 0, stream>>>(agg1, 0.f, (long long)N * F_MID);

    edge_max_kernel<<<cdivll(ET, T), T, 0, stream>>>(ei, a_s1, a_d1, m1, E, ET, H1);
    edge_exp_kernel<<<cdivll(ET, T), T, 0, stream>>>(ei, a_s1, a_d1, m1, expe1, den1, E, ET, H1);
    edge_agg_kernel<<<ET, 64, 0, stream>>>(ei, expe1, den1, xw1, agg1, E, H1, C1, F_MID, F_MID);

    elu_bias_f16_kernel<<<cdivll((long long)N * F_MID, T), T, 0, stream>>>(
        agg1, b1, hh, (long long)N * F_MID, F_MID);

    // ===================== Layer 2 =====================
    transpose_cvt_kernel<<<cdivll((long long)C2P * F_MID, T), T, 0, stream>>>(
        W2, w2t, F_MID, C2, C2P);

    gemm_wmma_f16_kernel<256, 48><<<(N + 15) / 16, 256, 0, stream>>>(
        hh, w2t, hw2, N);

    att_dot_kernel<<<cdivll(N, T), T, 0, stream>>>(hw2, as2v, ad2v, a_s2, a_d2, N, 1, C2, C2P);

    fill_f32_kernel<<<cdivll(N, T), T, 0, stream>>>(m2, -INFINITY, N);
    fill_f32_kernel<<<cdivll(N, T), T, 0, stream>>>(den2, 0.f, N);
    fill_f32_kernel<<<cdivll((long long)N * C2, T), T, 0, stream>>>(agg2, 0.f, (long long)N * C2);

    edge_max_kernel<<<cdivll(ET, T), T, 0, stream>>>(ei, a_s2, a_d2, m2, E, ET, 1);
    edge_exp_kernel<<<cdivll(ET, T), T, 0, stream>>>(ei, a_s2, a_d2, m2, expe2, den2, E, ET, 1);
    edge_agg_kernel<<<ET, 32, 0, stream>>>(ei, expe2, den2, hw2, agg2, E, 1, C2, C2P, C2);

    log_softmax_kernel<<<cdivll(N, T), T, 0, stream>>>(agg2, b2, out, N, C2);
}
